// ModulatedSparseTransformerCrossBlock_14465449853513
// MI455X (gfx1250) — matrix operand order repaired and checked by reading, not compiled, not run on hardware
//
#include <hip/hip_runtime.h>
#include <math.h>

typedef __attribute__((ext_vector_type(16))) __bf16 v16bf;
typedef __attribute__((ext_vector_type(8)))  float  v8f;

#define L_SEQ 8192
#define C_DIM 1024
#define H_NUM 16
#define D_DIM 64
#define LCTX  256

// A-operand k index for 16-bit WMMA A fragment (16x32): vgpr pair i, lane half.
__device__ __forceinline__ int a_k0(int i, int half) {
    return (i < 4) ? (2 * i + 8 * half) : (16 + 2 * (i - 4) + 8 * half);
}

// ---- CDNA5 async global->LDS copy (ASYNCcnt-tracked), per-lane addresses.
__device__ __forceinline__ void async_ld_b128(const void* g, unsigned lds_off) {
    asm volatile("global_load_async_to_lds_b128 %0, %1, off"
                 :: "v"(lds_off), "v"((unsigned long long)(size_t)g)
                 : "memory");
}
#define ASYNC_WAIT0() asm volatile("s_wait_asynccnt 0x0" ::: "memory")
#define ASYNC_WAIT4() asm volatile("s_wait_asynccnt 0x4" ::: "memory")

// ---------------------------------------------------------------- f32 -> bf16
__global__ void k_cvt(const float* __restrict__ s, __bf16* __restrict__ d, long n) {
    for (long i = (long)blockIdx.x * blockDim.x + threadIdx.x; i < n;
         i += (long)gridDim.x * blockDim.x)
        d[i] = (__bf16)s[i];
}

// ------------------------------------------------- m = silu(mod) @ W + b  [B,6C]
__global__ __launch_bounds__(256) void k_mod(const float* __restrict__ mod,
                                             const float* __restrict__ mw,
                                             const float* __restrict__ mb,
                                             float* __restrict__ outm) {
    __shared__ float s[1024];
    int b = blockIdx.y;
    int n = blockIdx.x * 256 + threadIdx.x;
    for (int i = threadIdx.x; i < 1024; i += 256) {
        float v = mod[b * 1024 + i];
        s[i] = v / (1.0f + __expf(-v));
    }
    __syncthreads();
    float acc = mb[n];
    for (int k = 0; k < 1024; ++k) acc += s[k] * mw[(long)k * 6144 + n];
    outm[(long)b * 6144 + n] = acc;
}

// ------------------- h = LN(x)*scale + shift  -> bf16 ; scale/shift from modm or vecs
__global__ __launch_bounds__(256) void k_lnmod(const float* __restrict__ x,
                                               __bf16* __restrict__ out,
                                               const float* __restrict__ modm,
                                               int sh_off, int sc_off,
                                               const float* __restrict__ gvec,
                                               const float* __restrict__ bvec) {
    int row = blockIdx.x;
    int b = row >> 13;  // row / L
    const float* xr = x + (long)row * C_DIM;
    int t = threadIdx.x;
    float4 v = ((const float4*)xr)[t];
    float s = v.x + v.y + v.z + v.w;
    float ss = v.x * v.x + v.y * v.y + v.z * v.z + v.w * v.w;
    for (int off = 16; off > 0; off >>= 1) {
        s  += __shfl_xor(s,  off, 32);
        ss += __shfl_xor(ss, off, 32);
    }
    __shared__ float rs_[8], rss_[8], mu_s, rstd_s;
    int wid = t >> 5, lane = t & 31;
    if (lane == 0) { rs_[wid] = s; rss_[wid] = ss; }
    __syncthreads();
    if (t == 0) {
        float S = 0.f, SS = 0.f;
        for (int i = 0; i < 8; ++i) { S += rs_[i]; SS += rss_[i]; }
        float mu = S * (1.0f / 1024.0f);
        float var = SS * (1.0f / 1024.0f) - mu * mu;
        mu_s = mu;
        rstd_s = rsqrtf(var + 1e-6f);
    }
    __syncthreads();
    float mu = mu_s, rstd = rstd_s;
    int c0 = t * 4;
    __bf16* orow = out + (long)row * C_DIM + c0;
    float xv[4] = {v.x, v.y, v.z, v.w};
#pragma unroll
    for (int j = 0; j < 4; ++j) {
        float sc, sh;
        if (modm) {
            const float* mrow = modm + (long)b * 6144;
            sc = 1.0f + mrow[sc_off + c0 + j];
            sh = mrow[sh_off + c0 + j];
        } else {
            sc = gvec[c0 + j];
            sh = bvec[c0 + j];
        }
        orow[j] = (__bf16)((xv[j] - mu) * rstd * sc + sh);
    }
}

// ------------------------------------------------------------ tiled WMMA GEMM
// out = act(A[MxK] @ W[KxN] + bias);  MODE 0: bf16  1: bf16+gelu
// MODE 2: f32 out = resid + val*gate (gate per (batch,col) from modm; in-place safe)
// Async double-buffered LDS pipeline: issue tile t+1, wait for tile t, compute.
#define BM 128
#define BN 128
#define BK 32
#define SA 40
#define SB 136

template <int MODE>
__global__ __launch_bounds__(256) void k_gemm(const __bf16* __restrict__ A,
                                              const __bf16* __restrict__ W,
                                              const float* __restrict__ bias,
                                              void* __restrict__ outp,
                                              const float* __restrict__ resid,
                                              const float* __restrict__ gates,
                                              int gate_off, int M, int N, int K) {
    __shared__ __bf16 As[2][BM * SA];
    __shared__ __bf16 Bs[2][BK * SB];
    int t = threadIdx.x, lane = t & 31, w = t >> 5;
    int wr = w & 3, wc = w >> 2;
    int half = lane >> 4, l16 = lane & 15;
    long rowBase = (long)blockIdx.x * BM;
    long colBase = (long)blockIdx.y * BN;

    v8f zf = {};
    v8f acc[2][4];
#pragma unroll
    for (int ms = 0; ms < 2; ++ms)
#pragma unroll
        for (int ns = 0; ns < 4; ++ns) acc[ms][ns] = zf;

    // async-issue one K-tile (4 x b128 per thread -> ASYNCcnt += 4 per wave)
    auto issue = [&](int buf, int kk) {
        {   // A tile: 128x32
            int r = t >> 1, seg = t & 1;
            const __bf16* g = A + (rowBase + r) * K + kk + seg * 16;
            unsigned l = (unsigned)(size_t)&As[buf][r * SA + seg * 16];
            async_ld_b128(g, l);
            async_ld_b128(g + 8, l + 16);
        }
        {   // B tile: 32x128
            int r = t >> 3, seg = t & 7;
            const __bf16* g = W + (long)(kk + r) * N + colBase + seg * 16;
            unsigned l = (unsigned)(size_t)&Bs[buf][r * SB + seg * 16];
            async_ld_b128(g, l);
            async_ld_b128(g + 8, l + 16);
        }
    };

    int T = K / BK;
    issue(0, 0);
    for (int tt = 0; tt < T; ++tt) {
        int cur = tt & 1;
        if (tt + 1 < T) {
            issue(cur ^ 1, (tt + 1) * BK);
            ASYNC_WAIT4();     // previous tile landed; next tile stays in flight
        } else {
            ASYNC_WAIT0();
        }
        __syncthreads();

        v16bf af[2], bfr[4];
#pragma unroll
        for (int ms = 0; ms < 2; ++ms) {
            const __bf16* ap = &As[cur][(wr * 32 + ms * 16 + l16) * SA];
#pragma unroll
            for (int i = 0; i < 8; ++i) {
                int k0 = a_k0(i, half);
                af[ms][2 * i]     = ap[k0];
                af[ms][2 * i + 1] = ap[k0 + 1];
            }
        }
#pragma unroll
        for (int ns = 0; ns < 4; ++ns) {
            const __bf16* bp = &Bs[cur][wc * 64 + ns * 16 + l16];
#pragma unroll
            for (int e = 0; e < 16; ++e) bfr[ns][e] = bp[(e < 8 ? 8 * half + e : 8 + 8 * half + e) * SB];
        }
#pragma unroll
        for (int ms = 0; ms < 2; ++ms)
#pragma unroll
            for (int ns = 0; ns < 4; ++ns)
                acc[ms][ns] = __builtin_amdgcn_wmma_f32_16x16x32_bf16(
                    false, af[ms], false, bfr[ns], (short)0, acc[ms][ns], false, false);
        __syncthreads();
    }

#pragma unroll
    for (int ms = 0; ms < 2; ++ms)
#pragma unroll
        for (int ns = 0; ns < 4; ++ns) {
            long col = colBase + wc * 64 + ns * 16 + l16;
            float bv = bias[col];
#pragma unroll
            for (int i = 0; i < 8; ++i) {
                long row = rowBase + wr * 32 + ms * 16 + i + 8 * half;
                float v = acc[ms][ns][i] + bv;
                if (MODE == 1) {
                    float u = 0.7978845608028654f * (v + 0.044715f * v * v * v);
                    v = 0.5f * v * (1.0f + tanhf(u));
                }
                if (MODE <= 1) {
                    ((__bf16*)outp)[row * (long)N + col] = (__bf16)v;
                } else {
                    float g = gates ? gates[(row >> 13) * 6144L + gate_off + col] : 1.0f;
                    ((float*)outp)[row * (long)N + col] =
                        resid[row * (long)N + col] + v * g;
                }
            }
        }
}

// --------------------------------------------------- flash attention (WMMA)
// SELF=1: Q = qkv buffer [B,L,3C] (q|k|v), window = 256 q-chunk, roll by SHIFT.
// SELF=0: Q = q buffer [B,L,C]; KV = [B,256,2C] (k|v). Out: bf16 [B,L,C].
template <int SELF>
__global__ __launch_bounds__(512) void k_attn(const __bf16* __restrict__ Q,
                                              const __bf16* __restrict__ KV,
                                              __bf16* __restrict__ Out) {
    __shared__ __bf16 Ks[128 * 72];
    __shared__ __bf16 Vs[128 * 72];
    __shared__ __bf16 Ps[256 * 40];  // 16 waves x (16 rows x 32 cols, pad 40)
    int t = threadIdx.x, lane = t & 31, w = t >> 5;
    int half = lane >> 4, l16 = lane & 15;
    int qc = blockIdx.x, h = blockIdx.y, b = blockIdx.z;

    // ---- Q fragments (16 rows per wave, D=64 -> two k-steps)
    int qrow = qc * 256 + w * 16 + l16;
    long ql = SELF ? ((qrow + 128) & (L_SEQ - 1)) : qrow;
    const __bf16* qptr =
        SELF ? (Q + ((long)b * L_SEQ + ql) * 3072 + h * 64)
             : (Q + ((long)b * L_SEQ + ql) * 1024 + h * 64);
    v16bf aq[2];
#pragma unroll
    for (int s = 0; s < 2; ++s)
#pragma unroll
        for (int i = 0; i < 8; ++i) {
            int k0 = s * 32 + a_k0(i, half);
            aq[s][2 * i]     = qptr[k0];
            aq[s][2 * i + 1] = qptr[k0 + 1];
        }

    v8f zf = {};
    v8f o[4];
#pragma unroll
    for (int i = 0; i < 4; ++i) o[i] = zf;
    float m8[8], l8[8];
#pragma unroll
    for (int i = 0; i < 8; ++i) { m8[i] = -1e30f; l8[i] = 0.0f; }

    const float scale = 0.125f;  // 1/sqrt(64)

    for (int ko = 0; ko < 2; ++ko) {
        __syncthreads();
        // async-load 128 keys of K and V into LDS (2 x b128 per thread per iter)
#pragma unroll
        for (int it = 0; it < 2; ++it) {
            int idx = t + it * 512;          // 0..1023 = 128 rows x 8 segs
            int r = idx >> 3, d0 = (idx & 7) * 8;
            int krow = ko * 128 + r;
            const __bf16 *kp, *vp;
            if (SELF) {
                long kl = ((long)qc * 256 + krow + 128) & (L_SEQ - 1);
                const __bf16* base = Q + ((long)b * L_SEQ + kl) * 3072 + h * 64;
                kp = base + 1024;
                vp = base + 2048;
            } else {
                const __bf16* base = KV + ((long)b * LCTX + krow) * 2048 + h * 64;
                kp = base;
                vp = base + 1024;
            }
            async_ld_b128(kp + d0, (unsigned)(size_t)&Ks[r * 72 + d0]);
            async_ld_b128(vp + d0, (unsigned)(size_t)&Vs[r * 72 + d0]);
        }
        ASYNC_WAIT0();
        __syncthreads();

        for (int c4 = 0; c4 < 4; ++c4) {
            int kb = c4 * 32;
            // scores: 16 q rows x 32 keys (two 16-col tiles)
            v8f st[2];
            st[0] = zf; st[1] = zf;
#pragma unroll
            for (int nt = 0; nt < 2; ++nt)
#pragma unroll
                for (int ks = 0; ks < 2; ++ks) {
                    v16bf bk;
#pragma unroll
                    for (int e = 0; e < 16; ++e)
                        bk[e] = Ks[(kb + nt * 16 + l16) * 72 + ks * 32 + (e < 8 ? 8 * half + e : 8 + 8 * half + e)];
                    st[nt] = __builtin_amdgcn_wmma_f32_16x16x32_bf16(
                        false, aq[ks], false, bk, (short)0, st[nt], false, false);
                }
            // online softmax per owned row, write P tile to LDS (bf16)
#pragma unroll
            for (int i = 0; i < 8; ++i) {
                float v0 = st[0][i] * scale, v1 = st[1][i] * scale;
                float mx = fmaxf(v0, v1);
                for (int off = 8; off > 0; off >>= 1)
                    mx = fmaxf(mx, __shfl_xor(mx, off, 16));
                float mn = fmaxf(m8[i], mx);
                float p0 = __expf(v0 - mn), p1 = __expf(v1 - mn);
                float rs = p0 + p1;
                for (int off = 8; off > 0; off >>= 1)
                    rs += __shfl_xor(rs, off, 16);
                float corr = __expf(m8[i] - mn);
                l8[i] = l8[i] * corr + rs;
                m8[i] = mn;
#pragma unroll
                for (int t4 = 0; t4 < 4; ++t4) o[t4][i] *= corr;
                int prow = w * 16 + i + 8 * half;
                Ps[prow * 40 + l16]      = (__bf16)p0;
                Ps[prow * 40 + 16 + l16] = (__bf16)p1;
            }
            // P (16x32) @ V (32x64)
            v16bf ap;
            const __bf16* pp = &Ps[(w * 16 + l16) * 40];
#pragma unroll
            for (int i = 0; i < 8; ++i) {
                int k0 = a_k0(i, half);
                ap[2 * i]     = pp[k0];
                ap[2 * i + 1] = pp[k0 + 1];
            }
#pragma unroll
            for (int t4 = 0; t4 < 4; ++t4) {
                v16bf bvv;
                const __bf16* vp2 = &Vs[kb * 72 + t4 * 16 + l16];
#pragma unroll
                for (int e = 0; e < 16; ++e) bvv[e] = vp2[(e < 8 ? 8 * half + e : 8 + 8 * half + e) * 72];
                o[t4] = __builtin_amdgcn_wmma_f32_16x16x32_bf16(
                    false, ap, false, bvv, (short)0, o[t4], false, false);
            }
        }
    }

    // ---- epilogue: normalize, (un)roll, store bf16
    int qlocal = qc * 256 + w * 16;
#pragma unroll
    for (int i = 0; i < 8; ++i) {
        int r = qlocal + i + 8 * half;
        long ol = SELF ? ((r + 128) & (L_SEQ - 1)) : r;
        float inv = 1.0f / l8[i];
        __bf16* op = Out + ((long)b * L_SEQ + ol) * 1024 + h * 64;
#pragma unroll
        for (int t4 = 0; t4 < 4; ++t4)
            op[t4 * 16 + l16] = (__bf16)(o[t4][i] * inv);
    }
}

// ------------------------------------------------------------------ launcher
extern "C" void kernel_launch(void* const* d_in, const int* in_sizes, int n_in,
                              void* d_out, int out_size, void* d_ws, size_t ws_size,
                              hipStream_t stream) {
    const float* x        = (const float*)d_in[0];
    const float* mod      = (const float*)d_in[1];
    const float* context  = (const float*)d_in[2];
    const float* mod_w    = (const float*)d_in[3];
    const float* mod_b    = (const float*)d_in[4];
    const float* norm2_g  = (const float*)d_in[5];
    const float* norm2_b  = (const float*)d_in[6];
    const float* qkv_w    = (const float*)d_in[7];
    const float* qkv_b    = (const float*)d_in[8];
    const float* sa_out_w = (const float*)d_in[9];
    const float* sa_out_b = (const float*)d_in[10];
    const float* q_w      = (const float*)d_in[11];
    const float* q_b      = (const float*)d_in[12];
    const float* kv_w     = (const float*)d_in[13];
    const float* kv_b     = (const float*)d_in[14];
    const float* ca_out_w = (const float*)d_in[15];
    const float* ca_out_b = (const float*)d_in[16];
    const float* w1       = (const float*)d_in[17];
    const float* b1       = (const float*)d_in[18];
    const float* w2       = (const float*)d_in[19];
    const float* b2       = (const float*)d_in[20];
    float* out = (float*)d_out;

    char* wsb = (char*)d_ws;
    size_t off = 0;
    auto alloc = [&](size_t bytes) {
        size_t cur = off;
        off += (bytes + 255) & ~(size_t)255;
        return cur;
    };
    const long NT = 32768;  // B*L
    __bf16* bigbuf = (__bf16*)(wsb + alloc((size_t)NT * 4096 * 2));
    __bf16* abuf   = (__bf16*)(wsb + alloc((size_t)NT * 1024 * 2));
    __bf16* bbuf   = (__bf16*)(wsb + alloc((size_t)NT * 1024 * 2));
    __bf16* kvbuf  = (__bf16*)(wsb + alloc((size_t)1024 * 2048 * 2));
    __bf16* ctxbf  = (__bf16*)(wsb + alloc((size_t)1024 * 1024 * 2));
    float*  modm   = (float*)(wsb + alloc((size_t)4 * 6144 * 4));
    __bf16* qkvw_bf = (__bf16*)(wsb + alloc((size_t)1024 * 3072 * 2));
    __bf16* saw_bf  = (__bf16*)(wsb + alloc((size_t)1024 * 1024 * 2));
    __bf16* qw_bf   = (__bf16*)(wsb + alloc((size_t)1024 * 1024 * 2));
    __bf16* kvw_bf  = (__bf16*)(wsb + alloc((size_t)1024 * 2048 * 2));
    __bf16* caw_bf  = (__bf16*)(wsb + alloc((size_t)1024 * 1024 * 2));
    __bf16* w1_bf   = (__bf16*)(wsb + alloc((size_t)1024 * 4096 * 2));
    __bf16* w2_bf   = (__bf16*)(wsb + alloc((size_t)4096 * 1024 * 2));

    // weight / context conversion to bf16
    k_cvt<<<2048, 256, 0, stream>>>(qkv_w,    qkvw_bf, 1024L * 3072);
    k_cvt<<<2048, 256, 0, stream>>>(sa_out_w, saw_bf,  1024L * 1024);
    k_cvt<<<2048, 256, 0, stream>>>(q_w,      qw_bf,   1024L * 1024);
    k_cvt<<<2048, 256, 0, stream>>>(kv_w,     kvw_bf,  1024L * 2048);
    k_cvt<<<2048, 256, 0, stream>>>(ca_out_w, caw_bf,  1024L * 1024);
    k_cvt<<<2048, 256, 0, stream>>>(w1,       w1_bf,   1024L * 4096);
    k_cvt<<<2048, 256, 0, stream>>>(w2,       w2_bf,   4096L * 1024);
    k_cvt<<<2048, 256, 0, stream>>>(context,  ctxbf,   1024L * 1024);

    // modulation m = silu(mod) @ mod_w + mod_b
    k_mod<<<dim3(24, 4), 256, 0, stream>>>(mod, mod_w, mod_b, modm);

    // ---- MSA branch
    k_lnmod<<<32768, 256, 0, stream>>>(x, abuf, modm, /*sh*/ 0, /*sc*/ 1024,
                                       nullptr, nullptr);
    k_gemm<0><<<dim3(256, 24), 256, 0, stream>>>(abuf, qkvw_bf, qkv_b, bigbuf,
                                                 nullptr, nullptr, 0, 32768, 3072, 1024);
    k_attn<1><<<dim3(32, 16, 4), 512, 0, stream>>>(bigbuf, bigbuf, bbuf);
    k_gemm<2><<<dim3(256, 8), 256, 0, stream>>>(bbuf, saw_bf, sa_out_b, out,
                                                x, modm, /*g_msa*/ 2048,
                                                32768, 1024, 1024);

    // ---- cross-attention branch
    k_lnmod<<<32768, 256, 0, stream>>>(out, abuf, nullptr, 0, 0, norm2_g, norm2_b);
    k_gemm<0><<<dim3(256, 8), 256, 0, stream>>>(abuf, qw_bf, q_b, bbuf,
                                                nullptr, nullptr, 0, 32768, 1024, 1024);
    k_gemm<0><<<dim3(8, 16), 256, 0, stream>>>(ctxbf, kvw_bf, kv_b, kvbuf,
                                               nullptr, nullptr, 0, 1024, 2048, 1024);
    k_attn<0><<<dim3(32, 16, 4), 512, 0, stream>>>(bbuf, kvbuf, abuf);
    k_gemm<2><<<dim3(256, 8), 256, 0, stream>>>(abuf, caw_bf, ca_out_b, out,
                                                out, nullptr, 0, 32768, 1024, 1024);

    // ---- MLP branch
    k_lnmod<<<32768, 256, 0, stream>>>(out, abuf, modm, /*sh*/ 3072, /*sc*/ 4096,
                                       nullptr, nullptr);
    k_gemm<1><<<dim3(256, 32), 256, 0, stream>>>(abuf, w1_bf, b1, bigbuf,
                                                 nullptr, nullptr, 0, 32768, 4096, 1024);
    k_gemm<2><<<dim3(256, 8), 256, 0, stream>>>(bigbuf, w2_bf, b2, out,
                                                out, modm, /*g_mlp*/ 5120,
                                                32768, 1024, 4096);
}
